// DGCNN6_homo_26018911879468
// MI455X (gfx1250) — matrix operand order repaired and checked by reading, not compiled, not run on hardware
//
#include <hip/hip_runtime.h>
#include <hip/hip_bf16.h>

// ---------------------------------------------------------------------------
// DGCNN (3x DynamicEdgeConv + homophily head) for gfx1250.
// All GEMM work via v_wmma_f32_16x16x32_f16 (f16 in, f32 acc).
// Fragments are stored pre-swizzled (fragment-major: chunk[lane][16 halves])
// so each WMMA operand is fed by contiguous 32B vector loads, not 16 scalar
// half loads. Weights are converted once per call into fragment-major f16.
// ---------------------------------------------------------------------------

typedef __attribute__((ext_vector_type(16))) _Float16 v16h;
typedef __attribute__((ext_vector_type(8)))  float    v8f;

static constexpr int Bg = 32;    // graphs
static constexpr int Nn = 2048;  // nodes per graph
static constexpr int BN = Bg * Nn;

__device__ __forceinline__ v8f wmma16(v16h a, v16h b, v8f c) {
  return __builtin_amdgcn_wmma_f32_16x16x32_f16(false, a, false, b,
                                                (short)0, c, false, false);
}

// Position (lane*16 + i) of logical element (row r, K-in-chunk kc) inside a
// 512-half A-fragment chunk (contiguous K order: lane = 16*(kc>>4) + r, i = kc&15).
__device__ __forceinline__ int fragPosA(int r, int kc) {
  int h = kc >> 4;
  return (h * 16 + (r & 15)) * 16 + (kc & 15);
}

__device__ __forceinline__ float leaky(float v) { return v >= 0.f ? v : 0.01f * v; }

// ---------------------------------------------------------------------------
__global__ void k_zero(float* p, int n) {
  int g = blockIdx.x * blockDim.x + threadIdx.x;
  if (g < n) p[g] = 0.f;
}

__global__ void k_build_xx(const float* __restrict__ x, const float* __restrict__ pos,
                           float* __restrict__ xx) {
  int g = blockIdx.x * blockDim.x + threadIdx.x;
  if (g < BN) {
    xx[g * 4 + 0] = x[g];
    xx[g * 4 + 1] = pos[g * 3 + 0];
    xx[g * 4 + 2] = pos[g * 3 + 1];
    xx[g * 4 + 3] = pos[g * 3 + 2];
  }
}

// Convert X[B*N,F] f32 -> padded f16 row-major [B*N, FP]
template <int F, int FP>
__global__ void k_tof16(const float* __restrict__ X, _Float16* __restrict__ Xh) {
  int g = blockIdx.x * blockDim.x + threadIdx.x;
  if (g < BN) {
    #pragma unroll 8
    for (int c = 0; c < FP; ++c)
      Xh[(size_t)g * FP + c] = (c < F) ? (_Float16)X[(size_t)g * F + c] : (_Float16)0.f;
  }
}

template <int F>
__global__ void k_sumsq(const float* __restrict__ X, float* __restrict__ S) {
  int g = blockIdx.x * blockDim.x + threadIdx.x;
  if (g < BN) {
    float s = 0.f;
    #pragma unroll 4
    for (int c = 0; c < F; ++c) { float v = X[(size_t)g * F + c]; s += v * v; }
    S[g] = s;
  }
}

// Weight f32 [K][Ncols] -> fragment-major f16: chunk (cb*nks+ks) holds
// B-fragment halves: lane = 16*(kc>>4) + (col&15), i = kc&15.
__global__ void k_wswizzle(const float* __restrict__ w, _Float16* __restrict__ wf,
                           int K, int Ncols) {
  int nks = (K + 31) / 32;
  int total = (Ncols / 16) * nks * 512;
  int e = blockIdx.x * blockDim.x + threadIdx.x;
  if (e >= total) return;
  int chunk = e / 512, pos = e % 512;
  int lane = pos / 16, i = pos % 16;
  int cb = chunk / nks, ks = chunk % nks;
  int kc = ((lane >> 4) << 4) + i;          // kB(i, half)
  int k = ks * 32 + kc;
  int col = cb * 16 + (lane & 15);
  wf[e] = (k < K) ? (_Float16)w[(size_t)k * Ncols + col] : (_Float16)0.f;
}

// ---------------------------------------------------------------------------
// Fused kNN: one block = one (graph, 16-row tile), 8 waves.
// d = s_i + s_j - 2*(X Xt) via WMMA; B-fragments are straight v16h loads from
// the padded f16 copy of X (K runs contiguously per lane); A tile is built
// once in LDS in fragment-major order.
// ---------------------------------------------------------------------------
template <int F, int FP, int KSEL, bool EXCL>
__global__ void k_knn(const _Float16* __restrict__ Xh, const float* __restrict__ S,
                      int* __restrict__ outIdx) {
  constexpr int NKS = FP / 32;
  extern __shared__ char smem_raw[];
  float*    dbuf = (float*)smem_raw;                  // [16][Nn] f32
  _Float16* ldsA = (_Float16*)(dbuf + 16 * Nn);       // NKS*512 halves
  float*    srow = (float*)(ldsA + NKS * 512);        // [16]

  const int rowTiles = Nn / 16;
  const int b  = blockIdx.x / rowTiles;
  const int rt = blockIdx.x % rowTiles;
  const int rowBase = rt * 16;
  const _Float16* Xg = Xh + (size_t)b * Nn * FP;
  const float*    Sg = S + (size_t)b * Nn;

  const int tid = threadIdx.x;
  for (int e = tid; e < 16 * FP; e += blockDim.x) {
    int r = e / FP, c = e % FP;
    ldsA[(c >> 5) * 512 + fragPosA(r, c & 31)] = Xg[(size_t)(rowBase + r) * FP + c];
  }
  if (tid < 16) srow[tid] = Sg[rowBase + tid];
  __syncthreads();

  const int wave = tid >> 5, lane = tid & 31;
  const int half = lane >> 4, lm = lane & 15;
  const v16h* av = (const v16h*)ldsA;

  for (int ct = wave; ct < Nn / 16; ct += 8) {
    const int colBase = ct * 16;
    const v16h* xv = (const v16h*)(Xg + (size_t)(colBase + lm) * FP);
    v8f acc = {0.f, 0.f, 0.f, 0.f, 0.f, 0.f, 0.f, 0.f};
    #pragma unroll
    for (int ks = 0; ks < NKS; ++ks) {
      v16h a  = av[ks * 32 + lane];
      v16h bf = xv[ks * 2 + half];
      acc = wmma16(a, bf, acc);
    }
    float scol = Sg[colBase + lm];
    #pragma unroll
    for (int v = 0; v < 8; ++v) {
      int m = v + 8 * half;                 // C/D: M = vgpr + 8*half, N = lane&15
      float d = srow[m] + scol - 2.0f * acc[v];
      if (EXCL && (colBase + lm) == (rowBase + m)) d = 1e9f;
      dbuf[m * Nn + colBase + lm] = d;
    }
  }
  __syncthreads();

  // top-KSEL: each wave owns 2 rows; float4 scan + wave32 shfl_xor reduction
  for (int rr = 0; rr < 2; ++rr) {
    const int r = wave * 2 + rr;
    float* drow = dbuf + r * Nn;
    const float4* drow4 = (const float4*)drow;
    int* orow = outIdx + (size_t)(b * Nn + rowBase + r) * KSEL;
    for (int s = 0; s < KSEL; ++s) {
      float best = 3e38f; int bidx = 0;
      for (int cc = lane; cc < Nn / 4; cc += 32) {
        float4 d4 = drow4[cc];
        if (d4.x < best) { best = d4.x; bidx = cc * 4 + 0; }
        if (d4.y < best) { best = d4.y; bidx = cc * 4 + 1; }
        if (d4.z < best) { best = d4.z; bidx = cc * 4 + 2; }
        if (d4.w < best) { best = d4.w; bidx = cc * 4 + 3; }
      }
      #pragma unroll
      for (int off = 16; off > 0; off >>= 1) {
        float od = __shfl_xor(best, off, 32);
        int   oi = __shfl_xor(bidx, off, 32);
        if (od < best || (od == best && oi < bidx)) { best = od; bidx = oi; }
      }
      if (lane == 0) { orow[s] = bidx; drow[bidx] = 3e38f; }
      // same-wave DS ops are in-order; store is visible to next pass's loads
    }
  }
}

// ---------------------------------------------------------------------------
// Fused EdgeConv MLP: one wave per 16 message rows; msg and h1 tiles kept in
// fragment-major LDS; weights read as fragment-major f16 vector loads.
// 'add' aggregation over k=5 via atomicAdd.
// ---------------------------------------------------------------------------
template <int IN, int H1>
__global__ void __launch_bounds__(32)
k_edgeconv(const float* __restrict__ X, const int* __restrict__ idx,
           const _Float16* __restrict__ w1f, const float* __restrict__ b1,
           const _Float16* __restrict__ w2f, const float* __restrict__ b2,
           float* __restrict__ Xout) {
  constexpr int MSG  = 2 * IN;
  constexpr int MSGP = (MSG < 32) ? 32 : MSG;
  constexpr int NK1  = MSGP / 32;          // k-chunks layer 1
  constexpr int NK2  = H1 / 32;            // k-chunks layer 2
  constexpr int OUT  = 64;
  __shared__ _Float16 msgs[NK1 * 512];
  __shared__ _Float16 h1s[NK2 * 512];

  const int msgBase = blockIdx.x * 16;
  const int lane = threadIdx.x, half = lane >> 4, lm = lane & 15;

  for (int e = lane; e < 16 * MSGP; e += 32) {
    int r = e / MSGP, c = e % MSGP;
    int row = msgBase + r;
    int g = row / 5;
    int bb = g / Nn;
    float v = 0.f;
    if (c < IN) {
      v = X[(size_t)g * IN + c];
    } else if (c < MSG) {
      int gj = bb * Nn + idx[row];
      int cc = c - IN;
      v = X[(size_t)gj * IN + cc] - X[(size_t)g * IN + cc];
    }
    msgs[(c >> 5) * 512 + fragPosA(r, c & 31)] = (_Float16)v;
  }
  __syncthreads();

  const v16h* mv = (const v16h*)msgs;
  const v16h* w1v = (const v16h*)w1f;
  for (int cb = 0; cb < H1 / 16; ++cb) {
    v8f acc = {0.f, 0.f, 0.f, 0.f, 0.f, 0.f, 0.f, 0.f};
    const int col = cb * 16 + lm;
    if (cb + 1 < H1 / 16)
      __builtin_prefetch(&w1v[(cb + 1) * NK1 * 32 + lane], 0, 0);
    #pragma unroll
    for (int ks = 0; ks < NK1; ++ks)
      acc = wmma16(mv[ks * 32 + lane], w1v[(cb * NK1 + ks) * 32 + lane], acc);
    float bias = b1[col];
    #pragma unroll
    for (int v = 0; v < 8; ++v) {
      int m = v + 8 * half;
      h1s[(col >> 5) * 512 + fragPosA(m, col & 31)] = (_Float16)leaky(acc[v] + bias);
    }
  }
  __syncthreads();

  const v16h* hv = (const v16h*)h1s;
  const v16h* w2v = (const v16h*)w2f;
  for (int cb = 0; cb < OUT / 16; ++cb) {
    v8f acc = {0.f, 0.f, 0.f, 0.f, 0.f, 0.f, 0.f, 0.f};
    const int col = cb * 16 + lm;
    if (cb + 1 < OUT / 16)
      __builtin_prefetch(&w2v[(cb + 1) * NK2 * 32 + lane], 0, 0);
    #pragma unroll
    for (int ks = 0; ks < NK2; ++ks)
      acc = wmma16(hv[ks * 32 + lane], w2v[(cb * NK2 + ks) * 32 + lane], acc);
    float bias = b2[col];
    #pragma unroll
    for (int v = 0; v < 8; ++v) {
      int m = v + 8 * half;
      int g = (msgBase + m) / 5;
      atomicAdd(&Xout[(size_t)g * OUT + col], leaky(acc[v] + bias));
    }
  }
}

// ---------------------------------------------------------------------------
// lin1: feat[.,196] @ w1(196x512) -> leaky -> @ w2(512x256) -> mean pool.
// One wave per 16 nodes; feat (K padded to 224) and hidden tiles in
// fragment-major LDS.
// ---------------------------------------------------------------------------
__global__ void __launch_bounds__(32)
k_lin1(const float* __restrict__ xx, const float* __restrict__ x1,
       const float* __restrict__ x2, const float* __restrict__ x3,
       const _Float16* __restrict__ w1f, const float* __restrict__ b1,
       const _Float16* __restrict__ w2f, const float* __restrict__ b2,
       float* __restrict__ pooled) {
  constexpr int K1 = 196, K1P = 224, NK1 = K1P / 32;   // 7
  constexpr int H = 512, NK2 = H / 32;                 // 16
  constexpr int O2 = 256;
  __shared__ _Float16 feat[NK1 * 512];
  __shared__ _Float16 hs[NK2 * 512];

  const int base = blockIdx.x * 16;
  const int lane = threadIdx.x, half = lane >> 4, lm = lane & 15;

  for (int e = lane; e < 16 * K1P; e += 32) {
    int r = e / K1P, c = e % K1P;
    int g = base + r;
    float v = 0.f;
    if      (c <   4) v = xx[(size_t)g *  4 + c];
    else if (c <  68) v = x1[(size_t)g * 64 + (c -   4)];
    else if (c < 132) v = x2[(size_t)g * 64 + (c -  68)];
    else if (c < 196) v = x3[(size_t)g * 64 + (c - 132)];
    feat[(c >> 5) * 512 + fragPosA(r, c & 31)] = (_Float16)v;
  }
  __syncthreads();

  const v16h* fv = (const v16h*)feat;
  const v16h* w1v = (const v16h*)w1f;
  for (int cb = 0; cb < H / 16; ++cb) {
    v8f acc = {0.f, 0.f, 0.f, 0.f, 0.f, 0.f, 0.f, 0.f};
    const int col = cb * 16 + lm;
    if (cb + 1 < H / 16)
      __builtin_prefetch(&w1v[(cb + 1) * NK1 * 32 + lane], 0, 0);
    #pragma unroll
    for (int ks = 0; ks < NK1; ++ks)
      acc = wmma16(fv[ks * 32 + lane], w1v[(cb * NK1 + ks) * 32 + lane], acc);
    float bias = b1[col];
    #pragma unroll
    for (int v = 0; v < 8; ++v) {
      int m = v + 8 * half;
      hs[(col >> 5) * 512 + fragPosA(m, col & 31)] = (_Float16)leaky(acc[v] + bias);
    }
  }
  __syncthreads();

  const v16h* hv = (const v16h*)hs;
  const v16h* w2v = (const v16h*)w2f;
  for (int cb = 0; cb < O2 / 16; ++cb) {
    v8f acc = {0.f, 0.f, 0.f, 0.f, 0.f, 0.f, 0.f, 0.f};
    const int col = cb * 16 + lm;
    if (cb + 1 < O2 / 16)
      __builtin_prefetch(&w2v[(cb + 1) * NK2 * 32 + lane], 0, 0);
    #pragma unroll
    for (int ks = 0; ks < NK2; ++ks)
      acc = wmma16(hv[ks * 32 + lane], w2v[(cb * NK2 + ks) * 32 + lane], acc);
    float bias = b2[col];
    #pragma unroll
    for (int v = 0; v < 8; ++v) {
      int m = v + 8 * half;
      int g = base + m;
      int b = g / Nn;
      atomicAdd(&pooled[b * O2 + col], (acc[v] + bias) * (1.0f / Nn));
    }
  }
}

// ---------------------------------------------------------------------------
__global__ void k_homophily(const float* __restrict__ xx, const int* __restrict__ hidx,
                            float* __restrict__ hsum) {
  int g = blockIdx.x * blockDim.x + threadIdx.x;
  if (g >= BN) return;
  int b = g / Nn;
  float xi0 = xx[g * 4 + 0], xi1 = xx[g * 4 + 1];
  float xi2 = xx[g * 4 + 2], xi3 = xx[g * 4 + 3];
  float c0 = 0.f, c1 = 0.f, c2 = 0.f, c3 = 0.f;
  for (int k = 0; k < 50; ++k) {
    int gj = b * Nn + hidx[(size_t)g * 50 + k];
    c0 += (xx[gj * 4 + 0] == xi0) ? 1.f : 0.f;
    c1 += (xx[gj * 4 + 1] == xi1) ? 1.f : 0.f;
    c2 += (xx[gj * 4 + 2] == xi2) ? 1.f : 0.f;
    c3 += (xx[gj * 4 + 3] == xi3) ? 1.f : 0.f;
  }
  const float inv = 1.0f / (float)(Nn * 50);
  atomicAdd(&hsum[b * 4 + 0], c0 * inv);
  atomicAdd(&hsum[b * 4 + 1], c1 * inv);
  atomicAdd(&hsum[b * 4 + 2], c2 * inv);
  atomicAdd(&hsum[b * 4 + 3], c3 * inv);
}

__global__ void k_final(const float* __restrict__ pooled, const float* __restrict__ hsum,
                        const float* __restrict__ mw1, const float* __restrict__ mb1,
                        const float* __restrict__ mw2, const float* __restrict__ mb2,
                        float* __restrict__ out) {
  __shared__ float z[256];
  const int b = blockIdx.x, t = threadIdx.x;
  float acc = mb1[t];
  for (int i = 0; i < 260; ++i) {
    float vi = (i < 256) ? pooled[b * 256 + i] : hsum[b * 4 + (i - 256)];
    acc += leaky(vi) * mw1[i * 256 + t];
  }
  z[t] = leaky(acc);
  __syncthreads();
  if (t < 3) {
    float o = mb2[t];
    for (int i = 0; i < 256; ++i) o += z[i] * mw2[i * 3 + t];
    out[b * 3 + t] = o;
  }
}

// ---------------------------------------------------------------------------
extern "C" void kernel_launch(void* const* d_in, const int* in_sizes, int n_in,
                              void* d_out, int out_size, void* d_ws, size_t ws_size,
                              hipStream_t stream) {
  (void)in_sizes; (void)n_in; (void)out_size; (void)ws_size;
  const float* x    = (const float*)d_in[0];
  const float* pos  = (const float*)d_in[1];
  // d_in[2] = batch (implicit in our [B,N] layout)
  const float* c1w1 = (const float*)d_in[3];  const float* c1b1 = (const float*)d_in[4];
  const float* c1w2 = (const float*)d_in[5];  const float* c1b2 = (const float*)d_in[6];
  const float* c2w1 = (const float*)d_in[7];  const float* c2b1 = (const float*)d_in[8];
  const float* c2w2 = (const float*)d_in[9];  const float* c2b2 = (const float*)d_in[10];
  const float* l1w1 = (const float*)d_in[11]; const float* l1b1 = (const float*)d_in[12];
  const float* l1w2 = (const float*)d_in[13]; const float* l1b2 = (const float*)d_in[14];
  const float* mw1  = (const float*)d_in[15]; const float* mb1  = (const float*)d_in[16];
  const float* mw2  = (const float*)d_in[17]; const float* mb2  = (const float*)d_in[18];
  float* out = (float*)d_out;

  // workspace bump allocation (float units; f16 regions counted as halves/2)
  float* ws = (float*)d_ws;
  size_t off = 0;
  float* xx     = ws + off; off += (size_t)BN * 4;
  float* sbuf   = ws + off; off += (size_t)BN;
  float* x1     = ws + off; off += (size_t)BN * 64;
  float* x2     = ws + off; off += (size_t)BN * 64;
  float* x3     = ws + off; off += (size_t)BN * 64;
  float* pooled = ws + off; off += (size_t)Bg * 256;
  float* hsum   = ws + off; off += (size_t)Bg * 4;
  int*   hidx   = (int*)(ws + off); off += (size_t)BN * 50;
  int*   eidx   = (int*)(ws + off); off += (size_t)BN * 5;
  _Float16* xh  = (_Float16*)(ws + off); off += (size_t)BN * 32;  // BN*64 halves
  // fragment-major f16 weights (sizes in halves; keep float offsets even)
  _Float16* c1w1f = (_Float16*)(ws + off); off += 2048   / 2;
  _Float16* c1w2f = (_Float16*)(ws + off); off += 4096   / 2;
  _Float16* c2w1f = (_Float16*)(ws + off); off += 16384  / 2;
  _Float16* c2w2f = (_Float16*)(ws + off); off += 8192   / 2;
  _Float16* l1w1f = (_Float16*)(ws + off); off += 114688 / 2;
  _Float16* l1w2f = (_Float16*)(ws + off); off += 131072 / 2;

  const int T = 256;
  auto blks = [](int n, int t) { return (n + t - 1) / t; };

  // init + weight conversion (deterministic, every call)
  k_build_xx<<<blks(BN, T), T, 0, stream>>>(x, pos, xx);
  k_zero<<<blks(BN * 64, T), T, 0, stream>>>(x1, BN * 64);
  k_zero<<<blks(BN * 64, T), T, 0, stream>>>(x2, BN * 64);
  k_zero<<<blks(BN * 64, T), T, 0, stream>>>(x3, BN * 64);
  k_zero<<<blks(Bg * 256, T), T, 0, stream>>>(pooled, Bg * 256);
  k_zero<<<1, T, 0, stream>>>(hsum, Bg * 4);
  k_wswizzle<<<blks(2048,   T), T, 0, stream>>>(c1w1, c1w1f,   8,  64);
  k_wswizzle<<<blks(4096,   T), T, 0, stream>>>(c1w2, c1w2f,  64,  64);
  k_wswizzle<<<blks(16384,  T), T, 0, stream>>>(c2w1, c2w1f, 128, 128);
  k_wswizzle<<<blks(8192,   T), T, 0, stream>>>(c2w2, c2w2f, 128,  64);
  k_wswizzle<<<blks(114688, T), T, 0, stream>>>(l1w1, l1w1f, 196, 512);
  k_wswizzle<<<blks(131072, T), T, 0, stream>>>(l1w2, l1w2f, 512, 256);

  const int knnGrid = Bg * (Nn / 16);
  const size_t shm32 = (size_t)16 * Nn * 4 + 1 * 512 * 2 + 64;  // FP=32
  const size_t shm64 = (size_t)16 * Nn * 4 + 2 * 512 * 2 + 64;  // FP=64

  // homophily graph (k=50, self-excluded) + per-channel homophily
  k_tof16<4, 32><<<blks(BN, T), T, 0, stream>>>(xx, xh);
  k_sumsq<4><<<blks(BN, T), T, 0, stream>>>(xx, sbuf);
  k_knn<4, 32, 50, true><<<knnGrid, 256, shm32, stream>>>(xh, sbuf, hidx);
  k_homophily<<<blks(BN, T), T, 0, stream>>>(xx, hidx, hsum);

  // EdgeConv 1: knn(xx, k=5, self included) -> MLP(8->64->64), sum over k
  k_knn<4, 32, 5, false><<<knnGrid, 256, shm32, stream>>>(xh, sbuf, eidx);
  k_edgeconv<4, 64><<<(BN * 5) / 16, 32, 0, stream>>>(xx, eidx, c1w1f, c1b1, c1w2f, c1b2, x1);

  // EdgeConv 2 (shared c2 weights): knn(x1) -> MLP(128->128->64)
  k_tof16<64, 64><<<blks(BN, T), T, 0, stream>>>(x1, xh);
  k_sumsq<64><<<blks(BN, T), T, 0, stream>>>(x1, sbuf);
  k_knn<64, 64, 5, false><<<knnGrid, 256, shm64, stream>>>(xh, sbuf, eidx);
  k_edgeconv<64, 128><<<(BN * 5) / 16, 32, 0, stream>>>(x1, eidx, c2w1f, c2b1, c2w2f, c2b2, x2);

  // EdgeConv 3 (shared c2 weights): knn(x2) -> MLP(128->128->64)
  k_tof16<64, 64><<<blks(BN, T), T, 0, stream>>>(x2, xh);
  k_sumsq<64><<<blks(BN, T), T, 0, stream>>>(x2, sbuf);
  k_knn<64, 64, 5, false><<<knnGrid, 256, shm64, stream>>>(xh, sbuf, eidx);
  k_edgeconv<64, 128><<<(BN * 5) / 16, 32, 0, stream>>>(x2, eidx, c2w1f, c2b1, c2w2f, c2b2, x3);

  // lin1 + mean pool
  k_lin1<<<BN / 16, 32, 0, stream>>>(xx, x1, x2, x3, l1w1f, l1b1, l1w2f, l1b2, pooled);

  // head MLP -> [B,3]
  k_final<<<Bg, 256, 0, stream>>>(pooled, hsum, mw1, mb1, mw2, mb2, out);
}
